// PEER_67723044324033
// MI455X (gfx1250) — compile-verified, matrix operand order checked by reading, not run on hardware
//
#include <hip/hip_runtime.h>
#include <hip/hip_bf16.h>
#include <math.h>

// ---------------------------------------------------------------------------
// PEER (product-key expert retrieval) for MI455X / gfx1250, wave32 + WMMA.
// Pipeline:
//   0) f32->f16 conversion (+transpose of w_q to N-major, keys to [h][p][k][d])
//   1) qproj GEMM  [2048x512]x[512x2048]  : TDM tile loads -> LDS -> WMMA
//   2) sim  GEMMs  per (p,h): [2048x128]x[128x256] via WMMA (A reused 4x)
//   3) top-16 of 256 per (p,token,head)    (wave32 shuffle argmax)
//   4) cartesian 16x16 -> top-16 + softmax (wave32 shuffle argmax)
//   5) gather down/up rows, dot+GELU+weighted sum, one workgroup per token
// ---------------------------------------------------------------------------

typedef __attribute__((ext_vector_type(16))) _Float16 v16h;
typedef __attribute__((ext_vector_type(8)))  _Float16 v8h;
typedef __attribute__((ext_vector_type(8)))  float    v8f;

#define TOKENS   2048   // B*N = 4*512
#define DIM      512
#define QCOLS    2048   // 2*H*DK
#define HEADS    8
#define DKEY     128
#define NKEYS    256
#define TOPK     16

#if defined(__has_builtin)
#if __has_builtin(__builtin_amdgcn_tensor_load_to_lds) && \
    __has_builtin(__builtin_amdgcn_s_wait_tensorcnt)
#define PEER_TDM 1
#endif
#endif
#ifndef PEER_TDM
#define PEER_TDM 0
#endif

// ---------------------------- stage 0: conversions -------------------------

__global__ __launch_bounds__(256) void peer_cvt_x(const float* __restrict__ x,
                                                  _Float16* __restrict__ xh, int n) {
    int i = blockIdx.x * 256 + threadIdx.x;
    if (i < n) xh[i] = (_Float16)x[i];
}

// w_q: [512 (k) x 2048 (n)] row-major -> wqt: [2048 (n) x 512 (k)] f16
__global__ __launch_bounds__(256) void peer_cvt_wq(const float* __restrict__ wq,
                                                   _Float16* __restrict__ wqt) {
    int i = blockIdx.x * 256 + threadIdx.x;          // 512*2048
    int k = i >> 11, n = i & 2047;
    wqt[(size_t)n * DIM + k] = (_Float16)wq[i];
}

// keys: [h][k][p][d] -> keysh: [h][p][k][d] f16 (so B-tile loads are contiguous)
__global__ __launch_bounds__(256) void peer_cvt_keys(const float* __restrict__ keys,
                                                     _Float16* __restrict__ keysh) {
    int i = blockIdx.x * 256 + threadIdx.x;          // 8*256*2*128 = 524288
    int d = i & 127;
    int k = (i >> 7) & 255;
    int p = (i >> 15) & 1;
    int h = i >> 16;
    keysh[(((size_t)(h * 2 + p) * NKEYS + k) * DKEY) + d] =
        (_Float16)keys[(((size_t)(h * NKEYS + k) * 2 + p) * DKEY) + d];
}

// ------------------- TDM descriptor: 2D f16 tile 64 rows x 32 cols ----------
// Source tensor: 2048 rows x 512 cols (row stride 512 elements, 2 B/elem).

#if PEER_TDM
typedef unsigned int v4u_t __attribute__((ext_vector_type(4)));
typedef int          v8i_t __attribute__((ext_vector_type(8)));
typedef int          v4i_t __attribute__((ext_vector_type(4)));

__device__ __forceinline__ void peer_tdm_load_tile(const _Float16* gsrc,
                                                   unsigned lds_off) {
    unsigned long long ga = (unsigned long long)(size_t)gsrc;
    // --- D# group 0 (ISA 8.3): count | lds_addr | global_addr[56:0] | type=2
    v4u_t g0;
    g0.x = 1u;                                             // count=1, user D#
    g0.y = lds_off;                                        // LDS byte address
    g0.z = (unsigned)ga;                                   // global_addr[31:0]
    g0.w = ((unsigned)(ga >> 32) & 0x01FFFFFFu) | (2u << 30); // [56:32] | type
    // --- D# group 1 (ISA 8.4)
    v8i_t g1;
    g1[0] = (int)(1u << 16);                               // data_size=1 (2 B)
    g1[1] = (int)((512u & 0xFFFFu) << 16);                 // tensor_dim0 lo16
    g1[2] = (int)((512u >> 16) | ((2048u & 0xFFFFu) << 16)); // d0 hi | d1 lo
    g1[3] = (int)((2048u >> 16) | (32u << 16));            // d1 hi | tile_dim0=32
    g1[4] = (int)64u;                                      // tile_dim1=64, dim2=0
    g1[5] = (int)512u;                                     // dim0_stride lo32
    g1[6] = 0;                                             // dim0_stride hi | d1_stride lo
    g1[7] = 0;                                             // d1_stride hi
    v4i_t g2 = {0, 0, 0, 0};                               // 2D: groups 2/3 unused
    v4i_t g3 = {0, 0, 0, 0};
    v8i_t g4 = {0, 0, 0, 0, 0, 0, 0, 0};                   // 6-arg toolchain extra group
    __builtin_amdgcn_tensor_load_to_lds(g0, g1, g2, g3, g4, 0);
}
#endif

// ---------------------------- stage 1: query projection --------------------
// 256 threads (8 waves) per block; block computes 64x64 of q.
// K in steps of 32: TDM stages A(64x32) and B(64x32) panels into LDS, then
// each wave does 2 WMMAs (16x32 of output, A fragment reused).
__global__ __launch_bounds__(256) void peer_qproj(const _Float16* __restrict__ xh,
                                                  const _Float16* __restrict__ wqt,
                                                  _Float16* __restrict__ qh) {
    __shared__ _Float16 As[64 * 32];
    __shared__ _Float16 Bs[64 * 32];
    int tid  = threadIdx.x;
    int w    = tid >> 5;
    int lane = tid & 31;
    int m0 = (blockIdx.x >> 5) * 64;      // 32 M-tiles
    int n0 = (blockIdx.x & 31) * 64;      // 32 N-tiles
    int mrow  = (w & 3) * 16 + (lane & 15);
    int nrow0 = (w >> 2) * 32 + (lane & 15);
    int kb    = (lane >> 4) * 16;
    v8f c0 = {}, c1 = {};
    for (int ks = 0; ks < 16; ++ks) {
        __syncthreads();                  // previous iteration's reads done
#if PEER_TDM
        if (tid < 32) {                   // wave 0 drives the Tensor Data Mover
            peer_tdm_load_tile(xh  + (size_t)m0 * DIM + ks * 32,
                               (unsigned)(size_t)(&As[0]));
            peer_tdm_load_tile(wqt + (size_t)n0 * DIM + ks * 32,
                               (unsigned)(size_t)(&Bs[0]));
            __builtin_amdgcn_s_wait_tensorcnt(0);
        }
#else
        {
            int idx = tid * 8;            // 2048 halves per panel / 256 threads
            int row = idx >> 5, col = idx & 31;
            *(v8h*)(As + idx) = *(const v8h*)(xh  + (size_t)(m0 + row) * DIM + ks * 32 + col);
            *(v8h*)(Bs + idx) = *(const v8h*)(wqt + (size_t)(n0 + row) * DIM + ks * 32 + col);
        }
#endif
        __syncthreads();
        v16h a  = *(const v16h*)(As + mrow * 32 + kb);
        v16h b0 = *(const v16h*)(Bs + nrow0 * 32 + kb);
        v16h b1 = *(const v16h*)(Bs + (nrow0 + 16) * 32 + kb);
        c0 = __builtin_amdgcn_wmma_f32_16x16x32_f16(false, a, false, b0,
                                                    (short)0, c0, false, false);
        c1 = __builtin_amdgcn_wmma_f32_16x16x32_f16(false, a, false, b1,
                                                    (short)0, c1, false, false);
    }
    int mo   = m0 + (w & 3) * 16 + (lane >> 4) * 8;
    int col0 = n0 + (w >> 2) * 32 + (lane & 15);
#pragma unroll
    for (int r = 0; r < 8; ++r) {
        qh[(size_t)(mo + r) * QCOLS + col0]      = (_Float16)c0[r];
        qh[(size_t)(mo + r) * QCOLS + col0 + 16] = (_Float16)c1[r];
    }
}

// ---------------------------- stage 2: sim GEMMs ---------------------------
// sim[p][t][h][key]; per (p,h): A=[2048x128] (q slice), B=[128x256] (keys).
// One wave computes 16 tokens x 64 keys: A fragment reused across 4 key tiles.
__global__ __launch_bounds__(256) void peer_sim(const _Float16* __restrict__ qh,
                                                const _Float16* __restrict__ keysh,
                                                float* __restrict__ sim) {
    int wid  = blockIdx.x * 8 + (threadIdx.x >> 5);   // 8192 waves
    int lane = threadIdx.x & 31;
    int kt4 = wid & 3;                                 // 64-key group
    int tt  = (wid >> 2) & 127;                        // token tile
    int ph  = wid >> 9;                                // p*8 + h, 0..15
    int p = ph >> 3, h = ph & 7;
    int t0 = tt * 16, n0 = kt4 * 64;
    int rrow = lane & 15;
    int kb   = (lane >> 4) * 16;
    const _Float16* arow  = qh + (size_t)(t0 + rrow) * QCOLS + ph * DKEY + kb;
    const _Float16* bbase = keysh + ((size_t)((h * 2 + p) * NKEYS + n0 + rrow)) * DKEY + kb;
    v8f acc[4] = {{}, {}, {}, {}};
#pragma unroll
    for (int ksv = 0; ksv < 4; ++ksv) {                // K = 128
        v16h a = *(const v16h*)(arow + ksv * 32);
#pragma unroll
        for (int nt = 0; nt < 4; ++nt) {
            v16h b = *(const v16h*)(bbase + (size_t)nt * 16 * DKEY + ksv * 32);
            acc[nt] = __builtin_amdgcn_wmma_f32_16x16x32_f16(false, a, false, b,
                                                             (short)0, acc[nt],
                                                             false, false);
        }
    }
    int to = t0 + (lane >> 4) * 8;
#pragma unroll
    for (int nt = 0; nt < 4; ++nt) {
        int key = n0 + nt * 16 + (lane & 15);
#pragma unroll
        for (int r = 0; r < 8; ++r)
            sim[((size_t)(p * TOKENS + to + r) * HEADS + h) * NKEYS + key] = acc[nt][r];
    }
}

// ---------------------------- stage 3: top-16 of 256 -----------------------
__global__ __launch_bounds__(256) void peer_topk1(const float* __restrict__ sim,
                                                  float* __restrict__ sx,
                                                  int* __restrict__ ix) {
    int row  = blockIdx.x * 8 + (threadIdx.x >> 5);
    int lane = threadIdx.x & 31;
    const float* sr = sim + (size_t)row * NKEYS;
    float vals[8];
#pragma unroll
    for (int j = 0; j < 8; ++j) vals[j] = sr[lane * 8 + j];
    unsigned removed = 0u;
    for (int s = 0; s < TOPK; ++s) {
        float bv = -3.4e38f; int bi = -1;
#pragma unroll
        for (int j = 0; j < 8; ++j) {
            int c = lane * 8 + j;
            if (!((removed >> j) & 1u) && (vals[j] > bv || (vals[j] == bv && c < bi))) {
                bv = vals[j]; bi = c;
            }
        }
        for (int off = 16; off > 0; off >>= 1) {
            float ov = __shfl_xor(bv, off, 32);
            int   oi = __shfl_xor(bi, off, 32);
            if (oi != -1 && (ov > bv || (ov == bv && (bi == -1 || oi < bi)))) { bv = ov; bi = oi; }
        }
        if (lane == (bi >> 3)) removed |= 1u << (bi & 7);
        if (lane == 0) { sx[(size_t)row * TOPK + s] = bv; ix[(size_t)row * TOPK + s] = bi; }
    }
}

// ------------------- stage 4: cartesian 16x16 top-16 + softmax -------------
__global__ __launch_bounds__(256) void peer_topk2(const float* __restrict__ sx,
                                                  const int* __restrict__ ix,
                                                  float* __restrict__ probs,
                                                  int* __restrict__ eidx) {
    int row  = blockIdx.x * 8 + (threadIdx.x >> 5);   // t*8 + h, 16384 rows
    int lane = threadIdx.x & 31;
    const float* sxr = sx + (size_t)row * TOPK;
    const float* syr = sx + (size_t)(16384 + row) * TOPK;
    const int*   ixr = ix + (size_t)row * TOPK;
    const int*   iyr = ix + (size_t)(16384 + row) * TOPK;
    float sxv = sxr[lane >> 1];
    float sy8[8];
#pragma unroll
    for (int j = 0; j < 8; ++j) sy8[j] = syr[(lane & 1) * 8 + j];
    unsigned removed = 0u;
    float m0 = 0.f, selv = 0.f; int selc = 0;
    for (int s = 0; s < TOPK; ++s) {
        float bv = -3.4e38f; int bc = -1;
#pragma unroll
        for (int j = 0; j < 8; ++j) {
            float v = sxv + sy8[j];
            int   c = lane * 8 + j;                    // == i*16+j global combo
            if (!((removed >> j) & 1u) && (v > bv || (v == bv && c < bc))) { bv = v; bc = c; }
        }
        for (int off = 16; off > 0; off >>= 1) {
            float ov = __shfl_xor(bv, off, 32);
            int   oc = __shfl_xor(bc, off, 32);
            if (oc != -1 && (ov > bv || (ov == bv && (bc == -1 || oc < bc)))) { bv = ov; bc = oc; }
        }
        if (s == 0) m0 = bv;
        if (lane == (bc >> 3)) removed |= 1u << (bc & 7);
        if (lane == s) { selv = bv; selc = bc; }
    }
    float e = (lane < TOPK) ? expf(selv - m0) : 0.f;
    float sum = e;
    for (int off = 16; off > 0; off >>= 1) sum += __shfl_xor(sum, off, 32);
    if (lane < TOPK) {
        int i = selc >> 4, j = selc & 15;
        probs[(size_t)row * TOPK + lane] = e / sum;
        eidx[(size_t)row * TOPK + lane]  = ixr[i] * NKEYS + iyr[j];
    }
}

// ---------------------- stage 5: expert gather + apply ---------------------
__global__ __launch_bounds__(256) void peer_experts(const float* __restrict__ x,
                                                    const float* __restrict__ down,
                                                    const float* __restrict__ up,
                                                    const float* __restrict__ probs,
                                                    const int* __restrict__ eidx,
                                                    float* __restrict__ out) {
    int t    = blockIdx.x;
    int tid  = threadIdx.x;
    int w    = tid >> 5;
    int lane = tid & 31;
    __shared__ float xs[DIM];
    __shared__ float accs[HEADS * DIM];
    xs[tid]       = x[(size_t)t * DIM + tid];
    xs[tid + 256] = x[(size_t)t * DIM + tid + 256];
    __syncthreads();

    float acc[16];
#pragma unroll
    for (int i = 0; i < 16; ++i) acc[i] = 0.f;

    const int base = (t * HEADS + w) * TOPK;
    for (int kk = 0; kk < TOPK; ++kk) {
        int   e  = eidx[base + kk];
        float pr = probs[base + kk];
        const float* dr = down + (size_t)e * DIM;
        const float* ur = up   + (size_t)e * DIM;
        if (kk + 1 < TOPK) {                          // hide gather latency
            int en = eidx[base + kk + 1];
            __builtin_prefetch(down + (size_t)en * DIM + lane * 16, 0, 0);
            __builtin_prefetch(up   + (size_t)en * DIM + lane * 16, 0, 0);
        }
        float dl = 0.f;
#pragma unroll
        for (int i = 0; i < 16; ++i) dl += xs[lane + 32 * i] * dr[lane + 32 * i];
        for (int off = 16; off > 0; off >>= 1) dl += __shfl_xor(dl, off, 32);
        float hv = 0.5f * dl * (1.0f + erff(dl * 0.70710678118654752f)) * pr;
#pragma unroll
        for (int i = 0; i < 16; ++i) acc[i] += hv * ur[lane + 32 * i];
    }
#pragma unroll
    for (int i = 0; i < 16; ++i) accs[w * DIM + lane + 32 * i] = acc[i];
    __syncthreads();
    float s0 = 0.f, s1 = 0.f;
#pragma unroll
    for (int w2 = 0; w2 < HEADS; ++w2) {
        s0 += accs[w2 * DIM + tid];
        s1 += accs[w2 * DIM + tid + 256];
    }
    out[(size_t)t * DIM + tid]       = s0;
    out[(size_t)t * DIM + tid + 256] = s1;
}

// ---------------------------------------------------------------------------

extern "C" void kernel_launch(void* const* d_in, const int* in_sizes, int n_in,
                              void* d_out, int out_size, void* d_ws, size_t ws_size,
                              hipStream_t stream) {
    (void)in_sizes; (void)n_in; (void)out_size; (void)ws_size;
    const float* x    = (const float*)d_in[0];   // [4,512,512]
    const float* w_q  = (const float*)d_in[1];   // [512,2048]
    const float* keys = (const float*)d_in[2];   // [8,256,2,128]
    const float* down = (const float*)d_in[3];   // [65536,512]
    const float* up   = (const float*)d_in[4];   // [65536,512]
    float* out = (float*)d_out;                  // [4,512,512]

    char* ws = (char*)d_ws;                      // 256B-aligned offsets
    _Float16* xh    = (_Float16*)(ws + 0);          // 2048*512   f16 =  2 MB
    _Float16* wqt   = (_Float16*)(ws + 2097152);    // 2048*512   f16 =  2 MB
    _Float16* keysh = (_Float16*)(ws + 4194304);    // 8*2*256*128 f16 = 1 MB
    _Float16* qh    = (_Float16*)(ws + 5242880);    // 2048*2048  f16 =  8 MB
    float*    sim   = (float*)   (ws + 13631488);   // 2*2048*8*256 f32 = 32 MB
    float*    sx    = (float*)   (ws + 47185920);   // 2*2048*8*16 f32 = 2 MB
    int*      ix    = (int*)     (ws + 49283072);   // 2*2048*8*16 i32 = 2 MB
    float*    probs = (float*)   (ws + 51380224);   // 2048*8*16  f32 =  1 MB
    int*      eidx  = (int*)     (ws + 52428800);   // 2048*8*16  i32 =  1 MB

    peer_cvt_x   <<<4096, 256, 0, stream>>>(x, xh, TOKENS * DIM);
    peer_cvt_wq  <<<4096, 256, 0, stream>>>(w_q, wqt);
    peer_cvt_keys<<<2048, 256, 0, stream>>>(keys, keysh);
    peer_qproj   <<<1024, 256, 0, stream>>>(xh, wqt, qh);      // TDM + WMMA
    peer_sim     <<<1024, 256, 0, stream>>>(qh, keysh, sim);   // WMMA, A reused 4x
    peer_topk1   <<<4096, 256, 0, stream>>>(sim, sx, ix);
    peer_topk2   <<<2048, 256, 0, stream>>>(sx, ix, probs, eidx);
    peer_experts <<<TOKENS, 256, 0, stream>>>(x, down, up, probs, eidx, out);
}
